// GatedFusion_69784628625675
// MI455X (gfx1250) — compile-verified, matrix operand order checked by reading, not run on hardware
//
#include <hip/hip_runtime.h>
#include <hip/hip_bf16.h>

#define DIMN 1024
#define KDIM 3072   // 3 * DIMN

typedef __attribute__((ext_vector_type(16))) __bf16 v16bf;
typedef __attribute__((ext_vector_type(8)))  float  v8f;

union Frag {
    v16bf v;
    uint4 u[2];
};

__device__ __forceinline__ unsigned short f2bf_bits(float x) {
    union { float f; unsigned u; } v; v.f = x;
    unsigned r = v.u + 0x7FFFu + ((v.u >> 16) & 1u);  // round-to-nearest-even
    return (unsigned short)(r >> 16);
}

// ---------------------------------------------------------------------------
// Kernel 1: per-row layernorm over concat(s,t,f) -> bf16 row; zero logits.
// One 256-thread block per row. Each thread owns 3 float4 (12 elems).
// ---------------------------------------------------------------------------
__global__ void ln_kernel(const float* __restrict__ s, const float* __restrict__ t,
                          const float* __restrict__ f, const float* __restrict__ lnw,
                          const float* __restrict__ lnb, unsigned short* __restrict__ xbf,
                          float* __restrict__ logits) {
    const int row  = blockIdx.x;
    const int tid  = threadIdx.x;
    const int lane = tid & 31;
    const int wid  = tid >> 5;

    float4 v[3];
    float sum = 0.f, sq = 0.f;
#pragma unroll
    for (int j = 0; j < 3; ++j) {
        const int col = (tid + 256 * j) * 4;           // 0..3068
        const float* src;
        if (col < DIMN)            src = s + (size_t)row * DIMN + col;
        else if (col < 2 * DIMN)   src = t + (size_t)row * DIMN + (col - DIMN);
        else                       src = f + (size_t)row * DIMN + (col - 2 * DIMN);
        v[j] = *(const float4*)src;
        sum += v[j].x + v[j].y + v[j].z + v[j].w;
        sq  += v[j].x * v[j].x + v[j].y * v[j].y + v[j].z * v[j].z + v[j].w * v[j].w;
    }

    __shared__ float rs[8], rq[8], bc[2];
#pragma unroll
    for (int off = 16; off; off >>= 1) {
        sum += __shfl_xor(sum, off, 32);
        sq  += __shfl_xor(sq,  off, 32);
    }
    if (lane == 0) { rs[wid] = sum; rq[wid] = sq; }
    __syncthreads();
    if (tid == 0) {
        float a = 0.f, b = 0.f;
#pragma unroll
        for (int i = 0; i < 8; ++i) { a += rs[i]; b += rq[i]; }
        const float mu  = a * (1.0f / (float)KDIM);
        const float var = b * (1.0f / (float)KDIM) - mu * mu;
        bc[0] = mu;
        bc[1] = rsqrtf(var + 1e-5f);
    }
    __syncthreads();
    const float mu = bc[0], rstd = bc[1];

#pragma unroll
    for (int j = 0; j < 3; ++j) {
        const int col = (tid + 256 * j) * 4;
        const float4 g = *(const float4*)(lnw + col);
        const float4 b = *(const float4*)(lnb + col);
        const float o0 = (v[j].x - mu) * rstd * g.x + b.x;
        const float o1 = (v[j].y - mu) * rstd * g.y + b.y;
        const float o2 = (v[j].z - mu) * rstd * g.z + b.z;
        const float o3 = (v[j].w - mu) * rstd * g.w + b.w;
        uint2 pk;
        pk.x = (unsigned)f2bf_bits(o0) | ((unsigned)f2bf_bits(o1) << 16);
        pk.y = (unsigned)f2bf_bits(o2) | ((unsigned)f2bf_bits(o3) << 16);
        *(uint2*)(xbf + (size_t)row * KDIM + col) = pk;
    }
    if (tid < 3) logits[(size_t)row * 3 + tid] = 0.f;
}

// ---------------------------------------------------------------------------
// Kernel 2: W1T[n][k] = bf16(W1[k][n]).  Tiled transpose via LDS.
// Grid (1024/32, 3072/32), block (32,8).
// ---------------------------------------------------------------------------
__global__ void w1t_kernel(const float* __restrict__ W1, unsigned short* __restrict__ w1t) {
    __shared__ float tile[32][33];
    const int n0 = blockIdx.x * 32;
    const int k0 = blockIdx.y * 32;
    const int tx = threadIdx.x, ty = threadIdx.y;
#pragma unroll
    for (int r = ty; r < 32; r += 8)
        tile[r][tx] = W1[(size_t)(k0 + r) * DIMN + n0 + tx];
    __syncthreads();
#pragma unroll
    for (int r = ty; r < 32; r += 8)
        w1t[(size_t)(n0 + r) * KDIM + k0 + tx] = f2bf_bits(tile[tx][r]);
}

// ---------------------------------------------------------------------------
// Kernel 3: h = relu(x @ W1 + b1) fused with partial logits += h @ W2.
// Block tile 256x128 (8 waves, 4x2). Wave tile 64x64 = 4x4 WMMA 16x16x32 bf16
// accumulators with register-level A/B reuse: 8 fragment loads (16 b128)
// feed 16 WMMAs per K=32 slab -> 1 b128 load per WMMA.
// Fragment lane layout per the ISA 16-bit A-matrix table:
//   lane<16 : row = lx, K = {kb..kb+7, kb+16..kb+23} with kb = 0
//   lane>=16: row = lx, same pattern with kb = 8
// ---------------------------------------------------------------------------
__global__ void __launch_bounds__(256, 1)
gemm_kernel(const unsigned short* __restrict__ xbf,
            const unsigned short* __restrict__ w1t,
            const float* __restrict__ b1,
            const float* __restrict__ W2,
            float* __restrict__ logits) {
    const int tid  = threadIdx.x;
    const int lane = tid & 31;
    const int wave = tid >> 5;          // 0..7
    const int wm   = wave & 3;          // 4 row-waves
    const int wn   = wave >> 2;         // 2 col-waves
    const int half = lane >> 4;
    const int lx   = lane & 15;

    const int rowBase = blockIdx.y * 256 + wm * 64;
    const int colBase = blockIdx.x * 128 + wn * 64;
    const int kb      = half * 8;       // lane-half K base inside a K=32 slab

    const unsigned short* ap[4];
    const unsigned short* bp[4];
#pragma unroll
    for (int i = 0; i < 4; ++i) {
        ap[i] = xbf + (size_t)(rowBase + i * 16 + lx) * KDIM + kb;
        bp[i] = w1t + (size_t)(colBase + i * 16 + lx) * KDIM + kb;
    }

    v8f acc[4][4] = {};

    for (int kk = 0; kk < KDIM; kk += 32) {
        Frag a[4], b[4];
#pragma unroll
        for (int i = 0; i < 4; ++i) {
            a[i].u[0] = *(const uint4*)(ap[i] + kk);
            a[i].u[1] = *(const uint4*)(ap[i] + kk + 16);
            b[i].u[0] = *(const uint4*)(bp[i] + kk);
            b[i].u[1] = *(const uint4*)(bp[i] + kk + 16);
        }
#pragma unroll
        for (int mi = 0; mi < 4; ++mi)
#pragma unroll
            for (int ni = 0; ni < 4; ++ni)
                acc[mi][ni] = __builtin_amdgcn_wmma_f32_16x16x32_bf16(
                    false, a[mi].v, false, b[ni].v, (short)0, acc[mi][ni], false, false);
    }

    // Epilogue: +b1, relu, multiply by W2 columns, reduce over N, atomic to logits.
    float b1n[4], w2x[4], w2y[4], w2z[4];
#pragma unroll
    for (int ni = 0; ni < 4; ++ni) {
        const int n = colBase + ni * 16 + lx;
        b1n[ni] = b1[n];
        w2x[ni] = W2[n * 3 + 0];
        w2y[ni] = W2[n * 3 + 1];
        w2z[ni] = W2[n * 3 + 2];
    }

#pragma unroll
    for (int mi = 0; mi < 4; ++mi) {
#pragma unroll
        for (int i = 0; i < 8; ++i) {
            float c0 = 0.f, c1 = 0.f, c2 = 0.f;
#pragma unroll
            for (int ni = 0; ni < 4; ++ni) {
                float hv = acc[mi][ni][i] + b1n[ni];
                hv = fmaxf(hv, 0.f);
                c0 += hv * w2x[ni];
                c1 += hv * w2y[ni];
                c2 += hv * w2z[ni];
            }
            // reduce across the 16 lanes of each half (halves map to rows i / i+8)
#pragma unroll
            for (int off = 1; off < 16; off <<= 1) {
                c0 += __shfl_xor(c0, off, 32);
                c1 += __shfl_xor(c1, off, 32);
                c2 += __shfl_xor(c2, off, 32);
            }
            if (lx == 0) {
                const int m = rowBase + mi * 16 + i + half * 8;
                atomicAdd(&logits[(size_t)m * 3 + 0], c0);
                atomicAdd(&logits[(size_t)m * 3 + 1], c1);
                atomicAdd(&logits[(size_t)m * 3 + 2], c2);
            }
        }
    }
}

// ---------------------------------------------------------------------------
// Kernel 4: softmax over 3 logits (+b2) and blend s,t,f.  One wave per row.
// ---------------------------------------------------------------------------
__global__ void fuse_kernel(const float* __restrict__ s, const float* __restrict__ t,
                            const float* __restrict__ f, const float* __restrict__ logits,
                            const float* __restrict__ b2, float* __restrict__ out) {
    const int wid  = threadIdx.x >> 5;
    const int lane = threadIdx.x & 31;
    const int row  = blockIdx.x * 8 + wid;

    const float l0 = logits[(size_t)row * 3 + 0] + b2[0];
    const float l1 = logits[(size_t)row * 3 + 1] + b2[1];
    const float l2 = logits[(size_t)row * 3 + 2] + b2[2];
    const float mx = fmaxf(l0, fmaxf(l1, l2));
    const float e0 = __expf(l0 - mx);
    const float e1 = __expf(l1 - mx);
    const float e2 = __expf(l2 - mx);
    const float inv = 1.0f / (e0 + e1 + e2);
    const float w0 = e0 * inv, w1 = e1 * inv, w2 = e2 * inv;

    const size_t base = (size_t)row * DIMN;
#pragma unroll
    for (int c = lane * 4; c < DIMN; c += 32 * 4) {
        const float4 sv = *(const float4*)(s + base + c);
        const float4 tv = *(const float4*)(t + base + c);
        const float4 fv = *(const float4*)(f + base + c);
        float4 o;
        o.x = w0 * sv.x + w1 * tv.x + w2 * fv.x;
        o.y = w0 * sv.y + w1 * tv.y + w2 * fv.y;
        o.z = w0 * sv.z + w1 * tv.z + w2 * fv.z;
        o.w = w0 * sv.w + w1 * tv.w + w2 * fv.w;
        *(float4*)(out + base + c) = o;
    }
}

// ---------------------------------------------------------------------------
extern "C" void kernel_launch(void* const* d_in, const int* in_sizes, int n_in,
                              void* d_out, int out_size, void* d_ws, size_t ws_size,
                              hipStream_t stream) {
    const float* s   = (const float*)d_in[0];
    const float* t   = (const float*)d_in[1];
    const float* f   = (const float*)d_in[2];
    const float* lnw = (const float*)d_in[3];
    const float* lnb = (const float*)d_in[4];
    const float* W1  = (const float*)d_in[5];
    const float* b1  = (const float*)d_in[6];
    const float* W2  = (const float*)d_in[7];
    const float* b2  = (const float*)d_in[8];
    float* out = (float*)d_out;

    const int B = in_sizes[0] / DIMN;   // 32768

    char* ws = (char*)d_ws;
    const size_t off_xbf  = 0;
    const size_t off_w1t  = off_xbf + (size_t)B * KDIM * sizeof(unsigned short);    // 201 MB
    const size_t off_logi = off_w1t + (size_t)DIMN * KDIM * sizeof(unsigned short); // +6 MB
    unsigned short* xbf    = (unsigned short*)(ws + off_xbf);
    unsigned short* w1t    = (unsigned short*)(ws + off_w1t);
    float*          logits = (float*)(ws + off_logi);

    // 1) layernorm -> bf16 activations (also zeroes logits)
    ln_kernel<<<B, 256, 0, stream>>>(s, t, f, lnw, lnb, xbf, logits);

    // 2) W1 -> W1T (bf16), contiguous along K for WMMA B-fragments
    w1t_kernel<<<dim3(DIMN / 32, KDIM / 32), dim3(32, 8), 0, stream>>>(W1, w1t);

    // 3) WMMA GEMM1 + relu + partial GEMM2 into logits
    gemm_kernel<<<dim3(DIMN / 128, B / 256), 256, 0, stream>>>(xbf, w1t, b1, W2, logits);

    // 4) softmax + gated blend
    fuse_kernel<<<B / 8, 256, 0, stream>>>(s, t, f, logits, b2, out);
}
